// DFD_76055280877843
// MI455X (gfx1250) — matrix operand order repaired and checked by reading, not compiled, not run on hardware
//
#include <hip/hip_runtime.h>
#include <cstdint>

typedef __attribute__((ext_vector_type(16))) _Float16 v16h;
typedef __attribute__((ext_vector_type(8)))  float    v8f;
typedef __attribute__((ext_vector_type(4)))  float    vf4;

#define NP      20352   /* padded anchors per batch = 128 * 159 */
#define NREAL   20267
#define BATCH   8
#define NMSC    1000
#define MAXDET  100

union HF32 { v16h v; vf4 q[2]; };

// --------- CDNA5 async global->LDS copy (bypasses VGPRs, ASYNCcnt) ---------
__device__ inline void async_copy16(unsigned lds_off, const _Float16* gptr) {
  asm volatile("global_load_async_to_lds_b128 %0, %1, off"
               :: "v"(lds_off), "v"(gptr) : "memory");
}
__device__ inline void wait_async() {
  asm volatile("s_wait_asynccnt 0x0" ::: "memory");
}
__device__ inline unsigned lds_off_of(const _Float16* p) {
  return (unsigned)(size_t)p;   // addrspace(3) ptr value == LDS byte offset
}

// --------- WMMA fragment loaders (per CDNA5 ISA 16-bit layouts) -----------
// A (16x32, MxK): lanes 0-15 row M=lane, K {0..7,16..23}; lanes 16-31 same row,
// K {8..15,24..31}. Two 16-byte chunks per lane.
__device__ inline v16h ldsA_frag(const _Float16* base, int stride, int lane, int k0) {
  int r  = lane & 15;
  int kb = k0 + ((lane >> 4) << 3);
  const _Float16* p = base + r * stride + kb;
  HF32 f;
  f.q[0] = *(const vf4*)(p);
  f.q[1] = *(const vf4*)(p + 16);
  return f.v;
}
// B (32x16, KxN) from W^T stored (N,K) row-major: lane 0-15 col N=lane holds
// K {0..7,16..23}; lanes 16-31 hold K {8..15,24..31} (matches A).
__device__ inline v16h ldsB_frag(const _Float16* base, int stride, int lane, int k0) {
  int n  = lane & 15;
  int kb = k0 + ((lane >> 4) << 3);
  const _Float16* p = base + n * stride + kb;
  HF32 f;
  f.q[0] = *(const vf4*)(p);
  f.q[1] = *(const vf4*)(p + 16);
  return f.v;
}

// --------- pack: (256, HW) f32 channel-major -> (NP, 256) f16 row-major ----
__global__ __launch_bounds__(256) void pack_level(const float* __restrict__ feat,
                                                  _Float16* __restrict__ X,
                                                  int HW, int n_off) {
  __shared__ _Float16 t[64 * 66];
  int pbase = blockIdx.x * 64;
  int cbase = blockIdx.y * 64;
  int tid = threadIdx.x;
  for (int i = tid; i < 64 * 64; i += 256) {
    int c = i >> 6, p = i & 63;
    float v = (pbase + p < HW) ? feat[(size_t)(cbase + c) * HW + pbase + p] : 0.f;
    t[c * 66 + p] = (_Float16)v;
  }
  __syncthreads();
  for (int i = tid; i < 64 * 64; i += 256) {
    int n = i >> 6, c = i & 63;
    if (pbase + n < HW)
      X[(size_t)(n_off + pbase + n) * 256 + cbase + c] = t[c * 66 + n];
  }
}

// --------- weight transpose+convert: W (256,N) f32 -> Wt (Npad,256) f16 ----
__global__ __launch_bounds__(256) void prep_w(const float* __restrict__ W,
                                              _Float16* __restrict__ Wt,
                                              int N, int Npad) {
  int i = blockIdx.x * 256 + threadIdx.x;
  if (i >= Npad * 256) return;
  int n = i >> 8, k = i & 255;
  float v = (n < N) ? W[k * N + n] : 0.f;
  Wt[n * 256 + k] = (_Float16)v;
}

// --------- GEMM: Y = act(X @ W + b), X (NP,256) f16, Wt (256,256) f16 ------
__global__ __launch_bounds__(256) void gemm256(const _Float16* __restrict__ X,
                                               const _Float16* __restrict__ Wt,
                                               const float* __restrict__ bias,
                                               _Float16* __restrict__ Y,
                                               int relu) {
  __shared__ _Float16 As[128 * 72];
  __shared__ _Float16 Bs[64 * 72];
  const int tid    = threadIdx.x;
  const int lane   = tid & 31;
  const int wv     = tid >> 5;
  const int rowblk = blockIdx.x * 128;
  const int colblk = blockIdx.y * 64;

  v8f acc[4];
  #pragma unroll
  for (int j = 0; j < 4; j++)
    #pragma unroll
    for (int e = 0; e < 8; e++) acc[j][e] = 0.f;

  for (int kb = 0; kb < 256; kb += 64) {
    // async global->LDS staging (ASYNCcnt-tracked, no VGPR round trip)
    #pragma unroll
    for (int i = tid; i < 128 * 8; i += 256) {
      int r = i >> 3, kc = (i & 7) << 3;
      async_copy16(lds_off_of(&As[r * 72 + kc]),
                   &X[(size_t)(rowblk + r) * 256 + kb + kc]);
    }
    #pragma unroll
    for (int i = tid; i < 64 * 8; i += 256) {
      int r = i >> 3, kc = (i & 7) << 3;
      async_copy16(lds_off_of(&Bs[r * 72 + kc]),
                   &Wt[(size_t)(colblk + r) * 256 + kb + kc]);
    }
    if (kb < 192)  // global_prefetch_b8 for the next K chunk
      __builtin_prefetch(&X[(size_t)(rowblk + (tid >> 1)) * 256 + kb + 64 + ((tid & 1) << 5)], 0, 1);
    wait_async();
    __syncthreads();
    #pragma unroll
    for (int kk = 0; kk < 64; kk += 32) {
      v16h a = ldsA_frag(&As[(wv * 16) * 72], 72, lane, kk);
      #pragma unroll
      for (int j = 0; j < 4; j++) {
        v16h b = ldsB_frag(&Bs[(j * 16) * 72], 72, lane, kk);
        acc[j] = __builtin_amdgcn_wmma_f32_16x16x32_f16(false, a, false, b,
                                                        (short)0, acc[j], false, false);
      }
    }
    __syncthreads();
  }

  const int n0   = lane & 15;
  const int rsel = (lane >> 4) << 3;
  #pragma unroll
  for (int j = 0; j < 4; j++) {
    int col = colblk + j * 16 + n0;
    float bv = bias[col];
    #pragma unroll
    for (int v = 0; v < 8; v++) {
      float val = acc[j][v] + bv;
      if (relu) val = fmaxf(val, 0.f);
      int row = rowblk + wv * 16 + v + rsel;
      Y[(size_t)row * 256 + col] = (_Float16)val;
    }
  }
}

// --------- fused layer-3 + score + decode ----------------------------------
__global__ __launch_bounds__(256) void head_kernel(const _Float16* __restrict__ Hc,
                                                   const _Float16* __restrict__ Hb,
                                                   const _Float16* __restrict__ Wc,
                                                   const float* __restrict__ bc,
                                                   const _Float16* __restrict__ Wb,
                                                   const float* __restrict__ bb,
                                                   float* __restrict__ scores,
                                                   float* __restrict__ boxes) {
  __shared__ _Float16 Asc[128 * 72];
  __shared__ _Float16 Asb[128 * 72];
  __shared__ _Float16 Bsc[80 * 72];
  __shared__ _Float16 Bsb[16 * 72];
  const int tid    = threadIdx.x;
  const int lane   = tid & 31;
  const int wv     = tid >> 5;
  const int rowblk = blockIdx.x * 128;

  v8f accc[5]; v8f accb;
  #pragma unroll
  for (int j = 0; j < 5; j++)
    #pragma unroll
    for (int e = 0; e < 8; e++) accc[j][e] = 0.f;
  #pragma unroll
  for (int e = 0; e < 8; e++) accb[e] = 0.f;

  for (int kb = 0; kb < 256; kb += 64) {
    #pragma unroll
    for (int i = tid; i < 128 * 8; i += 256) {
      int r = i >> 3, kc = (i & 7) << 3;
      async_copy16(lds_off_of(&Asc[r * 72 + kc]),
                   &Hc[(size_t)(rowblk + r) * 256 + kb + kc]);
      async_copy16(lds_off_of(&Asb[r * 72 + kc]),
                   &Hb[(size_t)(rowblk + r) * 256 + kb + kc]);
    }
    for (int i = tid; i < 80 * 8; i += 256) {
      int r = i >> 3, kc = (i & 7) << 3;
      async_copy16(lds_off_of(&Bsc[r * 72 + kc]),
                   &Wc[(size_t)r * 256 + kb + kc]);
    }
    if (tid < 16 * 8) {
      int r = tid >> 3, kc = (tid & 7) << 3;
      async_copy16(lds_off_of(&Bsb[r * 72 + kc]),
                   &Wb[(size_t)r * 256 + kb + kc]);
    }
    wait_async();
    __syncthreads();
    #pragma unroll
    for (int kk = 0; kk < 64; kk += 32) {
      v16h ac = ldsA_frag(&Asc[(wv * 16) * 72], 72, lane, kk);
      v16h ab = ldsA_frag(&Asb[(wv * 16) * 72], 72, lane, kk);
      #pragma unroll
      for (int j = 0; j < 5; j++) {
        v16h bt = ldsB_frag(&Bsc[(j * 16) * 72], 72, lane, kk);
        accc[j] = __builtin_amdgcn_wmma_f32_16x16x32_f16(false, ac, false, bt,
                                                         (short)0, accc[j], false, false);
      }
      v16h bb2 = ldsB_frag(&Bsb[0], 72, lane, kk);
      accb = __builtin_amdgcn_wmma_f32_16x16x32_f16(false, ab, false, bb2,
                                                    (short)0, accb, false, false);
    }
    __syncthreads();
  }

  const int n0    = lane & 15;
  const int rsel  = (lane >> 4) << 3;
  const int g     = lane & 16;
  float bbias = (n0 < 4) ? bb[n0] : 0.f;

  #pragma unroll
  for (int v = 0; v < 8; v++) {
    int n = rowblk + wv * 16 + v + rsel;
    // ---- class-max -> sigmoid score ----
    float mx = -3.4e38f;
    #pragma unroll
    for (int j = 0; j < 5; j++) mx = fmaxf(mx, accc[j][v] + bc[j * 16 + n0]);
    #pragma unroll
    for (int off = 8; off >= 1; off >>= 1)
      mx = fmaxf(mx, __shfl_xor(mx, off, 32));   // reduce within each 16-lane half
    if (n0 == 0)
      scores[n] = (n < NREAL) ? (1.f / (1.f + expf(-mx))) : -1e30f;
    // ---- box deltas -> decode ----
    float d  = accb[v] + bbias;
    float d0 = __shfl(d, g + 0, 32);
    float d1 = __shfl(d, g + 1, 32);
    float d2 = __shfl(d, g + 2, 32);
    float d3 = __shfl(d, g + 3, 32);
    if (n0 == 0 && n < NREAL) {
      int lvl, off, fh, fw;
      if      (n < 15200) { lvl = 0; off = 0;     fh = 100; fw = 152; }
      else if (n < 19000) { lvl = 1; off = 15200; fh = 50;  fw = 76;  }
      else if (n < 19950) { lvl = 2; off = 19000; fh = 25;  fw = 38;  }
      else if (n < 20197) { lvl = 3; off = 19950; fh = 13;  fw = 19;  }
      else                { lvl = 4; off = 20197; fh = 7;   fw = 10;  }
      int pos = n - off;
      int ro = pos / fw;
      int co = pos - ro * fw;
      float pcx = (co + 0.5f) / (float)fw;
      float pcy = (ro + 0.5f) / (float)fh;
      float pw  = 0.04f * (float)(1 << lvl);
      float cx = pcx + 1.5f * tanhf(d0) * pw;
      float cy = pcy + 1.5f * tanhf(d1) * pw;
      float w  = pw * expf(tanhf(d2));
      float h  = pw * expf(tanhf(d3));
      vf4 bxv;
      bxv.x = fminf(fmaxf(cx - 0.5f * w, 0.f), 1.f);
      bxv.y = fminf(fmaxf(cy - 0.5f * h, 0.f), 1.f);
      bxv.z = fminf(fmaxf(cx + 0.5f * w, 0.f), 1.f);
      bxv.w = fminf(fmaxf(cy + 0.5f * h, 0.f), 1.f);
      *(vf4*)&boxes[(size_t)n * 4] = bxv;
    }
  }
}

// --------- NMS: one workgroup (1024 thr) per batch -------------------------
__device__ inline void argmax_reduce(float& best, int& bi, float* s_rv, int* s_ri, int tid) {
  int lane = tid & 31, wv = tid >> 5;
  #pragma unroll
  for (int off = 16; off >= 1; off >>= 1) {
    float ov = __shfl_down(best, off, 32);
    int   oi = __shfl_down(bi,   off, 32);
    if (ov > best || (ov == best && oi < bi)) { best = ov; bi = oi; }
  }
  if (lane == 0) { s_rv[wv] = best; s_ri[wv] = bi; }
  __syncthreads();
  if (wv == 0) {
    best = s_rv[lane]; bi = s_ri[lane];
    #pragma unroll
    for (int off = 16; off >= 1; off >>= 1) {
      float ov = __shfl_down(best, off, 32);
      int   oi = __shfl_down(bi,   off, 32);
      if (ov > best || (ov == best && oi < bi)) { best = ov; bi = oi; }
    }
  }
  __syncthreads();
}

__global__ __launch_bounds__(1024) void nms_kernel(float* __restrict__ scores,
                                                   const float* __restrict__ boxes,
                                                   float* __restrict__ out) {
  __shared__ float s_cb[NMSC * 4];
  __shared__ float s_cs[NMSC];
  __shared__ int   s_keep[NMSC];
  __shared__ float s_ms[NMSC];
  __shared__ float s_rv[32];
  __shared__ int   s_ri[32];
  __shared__ int   s_widx;

  const int b   = blockIdx.x;
  const int tid = threadIdx.x;
  float*       smut = scores + (size_t)b * NP;    // destructively consumed, rebuilt each call
  const float* bxg  = boxes  + (size_t)b * NP * 4;

  // ---- top-1000 (sorted, stable ties -> lowest index) ----
  for (int it = 0; it < NMSC; it++) {
    float best = -3.4e38f; int bi = 1 << 30;
    for (int j = tid; j < NP; j += 1024) {
      float v = smut[j];
      if (v > best) { best = v; bi = j; }
    }
    argmax_reduce(best, bi, s_rv, s_ri, tid);
    if (tid == 0) { s_cs[it] = best; s_widx = bi; smut[bi] = -3.4e38f; }
    __syncthreads();
    if (tid < 4) s_cb[it * 4 + tid] = bxg[(size_t)s_widx * 4 + tid];
    __syncthreads();
  }

  // ---- greedy suppression ----
  if (tid < NMSC) s_keep[tid] = 1;
  __syncthreads();
  for (int i = 0; i < NMSC - 1; i++) {
    if (s_keep[i]) {
      float ix1 = s_cb[i * 4 + 0], iy1 = s_cb[i * 4 + 1];
      float ix2 = s_cb[i * 4 + 2], iy2 = s_cb[i * 4 + 3];
      float iar = fmaxf(ix2 - ix1, 0.f) * fmaxf(iy2 - iy1, 0.f);
      for (int j = i + 1 + tid; j < NMSC; j += 1024) {
        float jx1 = s_cb[j * 4 + 0], jy1 = s_cb[j * 4 + 1];
        float jx2 = s_cb[j * 4 + 2], jy2 = s_cb[j * 4 + 3];
        float jar = fmaxf(jx2 - jx1, 0.f) * fmaxf(jy2 - jy1, 0.f);
        float lx = fmaxf(ix1, jx1), ly = fmaxf(iy1, jy1);
        float rx = fminf(ix2, jx2), ry = fminf(iy2, jy2);
        float inter = fmaxf(rx - lx, 0.f) * fmaxf(ry - ly, 0.f);
        float iou = inter / fmaxf(iar + jar - inter, 1e-9f);
        if (iou > 0.5f) s_keep[j] = 0;
      }
    }
    __syncthreads();
  }

  // ---- top-100 of masked scores ----
  if (tid < NMSC) s_ms[tid] = s_keep[tid] ? s_cs[tid] : -1.0f;
  __syncthreads();
  for (int it = 0; it < MAXDET; it++) {
    float best = -3.4e38f; int bi = 1 << 30;
    if (tid < NMSC) { best = s_ms[tid]; bi = tid; }
    argmax_reduce(best, bi, s_rv, s_ri, tid);
    if (tid == 0) {
      s_widx = bi;
      out[((size_t)b * MAXDET + it) * 5 + 4] = best;
      s_ms[bi] = -3.4e38f;
    }
    __syncthreads();
    if (tid < 4) out[((size_t)b * MAXDET + it) * 5 + tid] = s_cb[s_widx * 4 + tid];
    __syncthreads();
  }
}

// ---------------------------------------------------------------------------
extern "C" void kernel_launch(void* const* d_in, const int* in_sizes, int n_in,
                              void* d_out, int out_size, void* d_ws, size_t ws_size,
                              hipStream_t stream) {
  (void)in_sizes; (void)n_in; (void)out_size; (void)ws_size;
  static const int HW[5]   = {15200, 3800, 950, 247, 70};
  static const int NOFF[5] = {0, 15200, 19000, 19950, 20197};

  // workspace carve (all offsets 256-B aligned)
  char* p = (char*)d_ws;
  _Float16* xb   = (_Float16*)p; p += (size_t)NP * 256 * 2;
  _Float16* hA   = (_Float16*)p; p += (size_t)NP * 256 * 2;
  _Float16* hB   = (_Float16*)p; p += (size_t)NP * 256 * 2;
  _Float16* Wt1c = (_Float16*)p; p += 256 * 256 * 2;
  _Float16* Wt2c = (_Float16*)p; p += 256 * 256 * 2;
  _Float16* Wt3c = (_Float16*)p; p += 80 * 256 * 2;
  _Float16* Wt1b = (_Float16*)p; p += 256 * 256 * 2;
  _Float16* Wt2b = (_Float16*)p; p += 256 * 256 * 2;
  _Float16* Wt3b = (_Float16*)p; p += 16 * 256 * 2;
  float* scoresW = (float*)p;    p += (size_t)BATCH * NP * 4;
  float* boxesW  = (float*)p;    p += (size_t)BATCH * NP * 16;

  // weights: transpose + f16 convert (once per call)
  prep_w<<<(256 * 256 + 255) / 256, 256, 0, stream>>>((const float*)d_in[5],  Wt1c, 256, 256);
  prep_w<<<(256 * 256 + 255) / 256, 256, 0, stream>>>((const float*)d_in[7],  Wt2c, 256, 256);
  prep_w<<<(80  * 256 + 255) / 256, 256, 0, stream>>>((const float*)d_in[9],  Wt3c, 80,  80);
  prep_w<<<(256 * 256 + 255) / 256, 256, 0, stream>>>((const float*)d_in[11], Wt1b, 256, 256);
  prep_w<<<(256 * 256 + 255) / 256, 256, 0, stream>>>((const float*)d_in[13], Wt2b, 256, 256);
  prep_w<<<(16  * 256 + 255) / 256, 256, 0, stream>>>((const float*)d_in[15], Wt3b, 4,   16);

  const dim3 gemm_grid(NP / 128, 4);
  for (int b = 0; b < BATCH; b++) {
    for (int l = 0; l < 5; l++) {
      const float* feat = (const float*)d_in[l] + (size_t)b * 256 * HW[l];
      pack_level<<<dim3((HW[l] + 63) / 64, 4), 256, 0, stream>>>(feat, xb, HW[l], NOFF[l]);
    }
    // cls MLP: x -> hA -> hB
    gemm256<<<gemm_grid, 256, 0, stream>>>(xb, Wt1c, (const float*)d_in[6],  hA, 1);
    gemm256<<<gemm_grid, 256, 0, stream>>>(hA, Wt2c, (const float*)d_in[8],  hB, 1);
    // box MLP: x -> hA -> xb (x no longer needed)
    gemm256<<<gemm_grid, 256, 0, stream>>>(xb, Wt1b, (const float*)d_in[12], hA, 1);
    gemm256<<<gemm_grid, 256, 0, stream>>>(hA, Wt2b, (const float*)d_in[14], xb, 1);
    // fused layer-3 + score + decode
    head_kernel<<<NP / 128, 256, 0, stream>>>(hB, xb, Wt3c, (const float*)d_in[10],
                                              Wt3b, (const float*)d_in[16],
                                              scoresW + (size_t)b * NP,
                                              boxesW  + (size_t)b * NP * 4);
  }
  nms_kernel<<<BATCH, 1024, 0, stream>>>(scoresW, boxesW, (float*)d_out);
}
